// onenet_odg_36069135352155
// MI455X (gfx1250) — matrix operand order repaired and checked by reading, not compiled, not run on hardware
//
#include <hip/hip_runtime.h>
#include <hip/hip_bf16.h>

// ---------------------------------------------------------------------------
// MI455X (gfx1250) implementation.
// All heavy contractions go through v_wmma_f32_16x16x32_bf16 (wave32 WMMA,
// fp32 accumulate). One generalized implicit-GEMM kernel (templated on the
// K-gather mode, B/ouput activations, and vectorized-B staging) serves every
// matmul / 1x1 conv / dilated conv / adjacency hop in the network.
//
// Cooperative LDS-staged, double-buffered design: per 32x256 block tile and
// 32-wide K-step, 256 threads gather A (2KB) and B (16KB) tiles once,
// converting fp32->bf16 during staging, written to LDS in exact WMMA fragment
// order. One barrier per K-step: while step i computes from buffer b, step
// i+1's global gathers land in buffer b^1 (gather latency hides under WMMA).
// VECB path: contiguous-B GEMMs (hops/skip/mw/end/y1) stage with coalesced
// float4 loads (8 global_load_b128 per thread/step instead of 32 b32).
// ---------------------------------------------------------------------------

typedef __attribute__((ext_vector_type(16))) __bf16 v16bf;
typedef __attribute__((ext_vector_type(8)))  float  v8f;

#define DEV_INLINE __device__ __forceinline__

DEV_INLINE float gelu_exact(float x) {
    return 0.5f * x * (1.0f + erff(x * 0.70710678118654752f));
}

template <int A>
DEV_INLINE float apply_act_t(float x) {
    if (A == 1) return fmaxf(x, 0.0f);             // relu
    if (A == 2) return tanhf(x);                   // tanh
    if (A == 3) return 1.0f / (1.0f + __expf(-x)); // sigmoid
    if (A == 4) return gelu_exact(x);              // gelu (exact, erf)
    return x;
}

DEV_INLINE unsigned pack_bf16(float a, float b) {
    unsigned short ua = __builtin_bit_cast(unsigned short, (__bf16)a);
    unsigned short ub = __builtin_bit_cast(unsigned short, (__bf16)b);
    return (unsigned)ua | ((unsigned)ub << 16);
}

// Generalized implicit-GEMM descriptor (all offsets fit in 32 bits).
//   D[m, j] = act( sum_k A[m,k] * Bgather(k,j) + bias[m] (+ D old) ) + addSrc
// B gather:  c = k / KDIV ; t = k % KDIV
//            j -> (j1, j2, j3) with sizes (.., n2, n3)
//            p2 = j2 + t*dil2 - off2   in [0, bound2) else 0
//            p3 = j3 + (t - pad3)*dil3 in [0, bound3) else 0
//            off = bOff + c*ks + j1*s1 + p2*s2 + p3*s3
struct GemmP {
    const float* A;
    const float* B;
    float*       D;
    const float* bias;    // per-row, may be null
    const float* addSrc;  // same layout as D, may be null (residual add)
    int M, K, N;          // N = number of output columns
    int aMs;              // A row stride (k contiguous). K even at all sites.
    int bOff, ks;
    int kdiv;             // dispatch only
    int n2, n3;
    int s1, s2, s3;
    int dil2, off2, bound2;
    int dil3, pad3, bound3;
    int bAct;             // dispatch only
    int outAct;           // dispatch only
    int accumulate;       // D += result
    int vecb;             // dispatch only: 1 = contiguous B (s3==1, n3%4==0)
};

// 256 threads = 8 waves (wave32). Block tile 32(M) x 256(N); waves 2(M)x4(N),
// each wave owns a 16x64 slice: 1 A fragment + 4 B fragments -> 4 WMMAs/step.
template <int KDIV, int BACT, int OACT, int VECB>
__global__ __launch_bounds__(256) void gemm_wmma_bf16(GemmP p) {
    // Double-buffered LDS, WMMA fragment order: [tile][lane(32)][8 dwords].
    __shared__ unsigned smA[2][2 * 256];    //  2KB/buf: two 16-row M-tiles
    __shared__ unsigned smB[2][16 * 256];   // 16KB/buf: sixteen 16-col N-tiles

    const int tid  = threadIdx.x;
    const int lane = tid & 31;
    const int wave = tid >> 5;

    // ---------------- staging-role precompute (once) ----------------
    // A: thread stages row sm = tid>>3, k-quad skl = (tid&7)*4.
    const int  sm    = tid >> 3;
    const int  skl   = (tid & 7) * 4;
    const int  grow  = blockIdx.y * 32 + sm;
    const bool rowOk = (grow < p.M);
    const float* __restrict__ ArowS = p.A + (unsigned)((rowOk ? grow : 0) * p.aMs);
    // A-frag layout: lane = (m&15) + 16*(kl>>4); dword = (kl&15)>>1
    const int aOff = (sm >> 4) * 256 +
                     ((sm & 15) + ((skl >> 4) << 4)) * 8 +
                     ((skl & 15) >> 1);

    const float* __restrict__ Bp = p.B;

    // B scalar-gather path: thread stages column col = tid.
    unsigned bAddrT[3]; bool bOkT[3];
    const int bColOff  = (tid >> 4) * 256;
    const int bLaneLow = tid & 15;
    if constexpr (!VECB) {
        const int  j   = blockIdx.x * 256 + tid;
        const bool jok = (j < p.N);
        const int  jj  = jok ? j : 0;
        const int  j1  = jj / (p.n2 * p.n3);
        const int  r   = jj - j1 * (p.n2 * p.n3);
        const int  j2  = r / p.n3;
        const int  j3  = r - j2 * p.n3;
#pragma unroll
        for (int t = 0; t < 3; ++t) {
            const int tt = (t < KDIV) ? t : 0;
            const int p2 = j2 + tt * p.dil2 - p.off2;
            const int p3 = j3 + (tt - p.pad3) * p.dil3;
            bOkT[t] = jok && ((unsigned)p2 < (unsigned)p.bound2) &&
                             ((unsigned)p3 < (unsigned)p.bound3);
            bAddrT[t] = (unsigned)(p.bOff + j1 * p.s1 + p2 * p.s2 + p3 * p.s3);
        }
    }

    // B vectorized path: thread stages one k-row (rk) x 32 columns (8 quads).
    unsigned bQuadAddr[8]; bool bQuadOk[8];
    const int rk = tid >> 3;          // local k-row
    const int cg = (tid & 7) * 32;    // first of 32 local columns
    if constexpr (VECB) {
#pragma unroll
        for (int q = 0; q < 8; ++q) {
            const int  jq = blockIdx.x * 256 + cg + 4 * q;
            const bool ok = (jq < p.N);     // n3 % 4 == 0 at all VECB sites
            const int  jjq = ok ? jq : 0;
            const int  j1q = jjq / (p.n2 * p.n3);
            const int  rq  = jjq - j1q * (p.n2 * p.n3);
            const int  j2q = rq / p.n3;
            const int  j3q = rq - j2q * p.n3;
            bQuadOk[q]   = ok;
            bQuadAddr[q] = (unsigned)(p.bOff + j1q * p.s1 + j2q * p.s2 + j3q);
        }
    }

    // ---------------- compute-role precompute ----------------
    const int half  = lane >> 4;
    const int l16   = lane & 15;
    const int waveM = wave & 1;
    const int waveN = wave >> 1;
    const int aReadOff = waveM * 256 + lane * 8;
    const int bReadOff = waveN * 4 * 256 + lane * 8;

    const v8f vzero = {0.f, 0.f, 0.f, 0.f, 0.f, 0.f, 0.f, 0.f};
    v8f acc[4];
#pragma unroll
    for (int s = 0; s < 4; ++s) acc[s] = vzero;

    // Staging helper (emitted once; uniform `if` around call site).
    auto stage = [&](int k0, int buf) {
        // ---- A: two aligned float2 loads -> one ds_store_b64 ----
        {
            const int  ka   = k0 + skl;
            const bool ok01 = rowOk && (ka < p.K);      // K even, ka even
            const bool ok23 = rowOk && (ka + 2 < p.K);
            const float2 x01 = *(const float2*)(ArowS + (ok01 ? ka : 0));
            const float2 x23 = *(const float2*)(ArowS + (ok23 ? ka + 2 : 0));
            uint2 pk;
            pk.x = pack_bf16(ok01 ? x01.x : 0.f, ok01 ? x01.y : 0.f);
            pk.y = pack_bf16(ok23 ? x23.x : 0.f, ok23 ? x23.y : 0.f);
            *(uint2*)&smA[buf][aOff] = pk;
        }
        if constexpr (VECB) {
            // ---- B: 8 coalesced float4 loads -> 32 ds_store_b16 ----
            const int  kk  = k0 + rk;
            const bool kOk = (kk < p.K);
            const unsigned kOff = (unsigned)(kk * p.ks);
            __bf16* const smBh = (__bf16*)&smB[buf][0];
            const int laneBase = ((rk >> 4) << 4);
            const int hw = rk & 15;
#pragma unroll
            for (int q = 0; q < 8; ++q) {
                const bool ok = bQuadOk[q] && kOk;
                const float4 xv =
                    *(const float4*)(Bp + (ok ? (bQuadAddr[q] + kOff) : 0u));
                float xs[4] = {xv.x, xv.y, xv.z, xv.w};
#pragma unroll
                for (int u = 0; u < 4; ++u) {
                    float x = ok ? xs[u] : 0.0f;
                    if (BACT) x = apply_act_t<BACT>(x);
                    const int c = cg + 4 * q + u;
                    smBh[(c >> 4) * 512 + ((c & 15) + laneBase) * 16 + hw] =
                        (__bf16)x;
                }
            }
        } else {
            // ---- B: 32 gathered elements -> 16 ds_store_b32 ----
#pragma unroll
            for (int w = 0; w < 16; ++w) {
                float xv[2];
#pragma unroll
                for (int e = 0; e < 2; ++e) {
                    const int k = k0 + 2 * w + e;  // k0 even -> tap parity folds
                    int c, t;
                    if (KDIV == 1)      { c = k;      t = 0;         }
                    else if (KDIV == 2) { c = k >> 1; t = k & 1;     }
                    else                { c = k / 3;  t = k - c * 3; }
                    const unsigned base =
                        (KDIV == 1) ? bAddrT[0]
                      : (KDIV == 2) ? (t ? bAddrT[1] : bAddrT[0])
                      : (t == 0 ? bAddrT[0] : (t == 1 ? bAddrT[1] : bAddrT[2]));
                    const bool okT =
                        (KDIV == 1) ? bOkT[0]
                      : (KDIV == 2) ? (t ? bOkT[1] : bOkT[0])
                      : (t == 0 ? bOkT[0] : (t == 1 ? bOkT[1] : bOkT[2]));
                    const bool ok = okT && (k < p.K);
                    unsigned a = base + (unsigned)(c * p.ks);
                    a = ok ? a : 0u;           // clamped: offset 0 is mapped
                    float x = Bp[a];
                    x = ok ? x : 0.0f;
                    if (BACT) x = apply_act_t<BACT>(x);
                    xv[e] = x;
                }
                const int kl = 2 * w;
                smB[buf][bColOff + (bLaneLow + ((kl >> 4) << 4)) * 8 +
                         ((kl & 15) >> 1)] = pack_bf16(xv[0], xv[1]);
            }
        }
    };

    // Prologue: fill buffer 0; then one barrier per K-step (double buffer).
    stage(0, 0);
    int buf = 0;
    for (int k0 = 0; k0 < p.K; k0 += 32) {
        __syncthreads();
        if (k0 + 32 < p.K) stage(k0 + 32, buf ^ 1);  // gathers overlap WMMAs

        union Frag { uint4 q[2]; v16bf v; };
        Frag ua, ub[4];
        {
            const unsigned* ap = &smA[buf][aReadOff];
            ua.q[0] = *(const uint4*)(ap);
            ua.q[1] = *(const uint4*)(ap + 4);
        }
#pragma unroll
        for (int s = 0; s < 4; ++s) {
            const unsigned* bp = &smB[buf][bReadOff + s * 256];
            ub[s].q[0] = *(const uint4*)(bp);
            ub[s].q[1] = *(const uint4*)(bp + 4);
        }
#pragma unroll
        for (int s = 0; s < 4; ++s)
            acc[s] = __builtin_amdgcn_wmma_f32_16x16x32_bf16(
                false, ua.v, false, ub[s].v, (short)0, acc[s], false, false);
        buf ^= 1;
    }

    // Branchless epilogue: pointer-select dummies keep all loads valid.
    const int m0 = blockIdx.y * 32 + waveM * 16;
    const int n0 = blockIdx.x * 256 + waveN * 64;
    const bool hasB   = (p.bias   != nullptr);
    const bool hasAdd = (p.addSrc != nullptr);
    const bool hasAcc = (p.accumulate != 0);
    const float* __restrict__ bpb = hasB   ? p.bias   : p.D;
    const float* __restrict__ asp = hasAdd ? p.addSrc : p.D;
#pragma unroll
    for (int v = 0; v < 8; ++v) {
        const int m = m0 + v + (half << 3);
        if (m >= p.M) continue;
        float bi = bpb[m];
        bi = hasB ? bi : 0.0f;
#pragma unroll
        for (int s = 0; s < 4; ++s) {
            const int n = n0 + s * 16 + l16;
            if (n >= p.N) continue;
            const unsigned idx = (unsigned)m * (unsigned)p.N + (unsigned)n;
            float dv = p.D[idx];
            float av = asp[idx];
            float val = acc[s][v] + (hasAcc ? dv : 0.0f) + bi;
            if (OACT) val = apply_act_t<OACT>(val);
            val += hasAdd ? av : 0.0f;
            p.D[idx] = val;
        }
    }
}

// ------------------------------- glue kernels ------------------------------

__global__ void ew_mul_k(float* __restrict__ a, const float* __restrict__ b, long n) {
    long i = (long)blockIdx.x * blockDim.x + threadIdx.x;
    if (i < n) a[i] *= b[i];
}

// acc (=|+=) t1[b*4+kk] * src ; tensors laid out [C=32, B=32, L, 512]
__global__ void axpy_b_k(float* __restrict__ acc, const float* __restrict__ src,
                         const float* __restrict__ t1, int kk, long LN, long n,
                         int init) {
    long i = (long)blockIdx.x * blockDim.x + threadIdx.x;
    if (i < n) {
        int b = (int)((i / LN) & 31);
        float s = t1[b * 4 + kk] * src[i];
        acc[i] = init ? s : (acc[i] + s);
    }
}

// io[c,b,l,n] += res[c,b,l+dOff,n]  (io has Lout, res has Lin)
__global__ void add_res_k(float* __restrict__ io, const float* __restrict__ res,
                          int Lout, int Lin, int dOff, long n) {
    long i = (long)blockIdx.x * blockDim.x + threadIdx.x;
    if (i < n) {
        int nn  = (int)(i & 511);
        long r  = i >> 9;
        int l   = (int)(r % Lout);
        long cb = r / Lout;  // c*32 + b
        io[i] += res[(cb * Lin + (l + dOff)) * 512 + nn];
    }
}

// dst[b*6144 + o*512 + n] = src[o*16384 + b*512 + n]  (o<12, b<32, n<512)
__global__ void ycopy_k(float* __restrict__ dst, const float* __restrict__ src) {
    int i = blockIdx.x * blockDim.x + threadIdx.x;
    if (i < 12 * 32 * 512) {
        int n = i & 511;
        int b = (i >> 9) & 31;
        int o = i >> 14;
        dst[b * 6144 + o * 512 + n] = src[o * 16384 + b * 512 + n];
    }
}

// adp = softmax(relu(nv1 @ nv2), axis=1); one row per block.
__global__ __launch_bounds__(512) void adp_kernel(const float* __restrict__ nv1,
                                                  const float* __restrict__ nv2,
                                                  float* __restrict__ adp) {
    int n = blockIdx.x, m = threadIdx.x;
    float acc = 0.f;
    for (int k = 0; k < 10; ++k) acc += nv1[n * 10 + k] * nv2[k * 512 + m];
    acc = fmaxf(acc, 0.f);
    __shared__ float red[512];
    red[m] = acc; __syncthreads();
    for (int s = 256; s > 0; s >>= 1) { if (m < s) red[m] = fmaxf(red[m], red[m + s]); __syncthreads(); }
    float mx = red[0]; __syncthreads();
    float e = __expf(acc - mx);
    red[m] = e; __syncthreads();
    for (int s = 256; s > 0; s >>= 1) { if (m < s) red[m] += red[m + s]; __syncthreads(); }
    adp[n * 512 + m] = e / red[0];
}

// timestamp embedding + layernorm + reduce + t1 head; one block (16 lanes) per b
__global__ void t1_kernel(const int* __restrict__ stamp, const float* __restrict__ semb,
                          const float* __restrict__ pemb, const float* __restrict__ lng,
                          const float* __restrict__ lnb, const float* __restrict__ rw,
                          const float* __restrict__ t1w, const float* __restrict__ t1b,
                          float* __restrict__ t1out) {
    int b = blockIdx.x, e = threadIdx.x;  // e in [0,16)
    float pe = 0.f;
    for (int t = 0; t < 12; ++t) {
        int s = stamp[b * 12 + t];
        float v = semb[s * 16 + e] + pemb[t * 16 + e];
        float sum = v;
        for (int o = 8; o > 0; o >>= 1) sum += __shfl_xor(sum, o, 16);
        float mu = sum * (1.f / 16.f);
        float d = v - mu;
        float sq = d * d;
        for (int o = 8; o > 0; o >>= 1) sq += __shfl_xor(sq, o, 16);
        float tn = d * rsqrtf(sq * (1.f / 16.f) + 1e-6f) * lng[e] + lnb[e];
        pe += tn * rw[t];
    }
    for (int k = 0; k < 4; ++k) {
        float part = pe * t1w[k * 16 + e];
        for (int o = 8; o > 0; o >>= 1) part += __shfl_xor(part, o, 16);
        if (e == 0) t1out[b * 4 + k] = part + t1b[k];
    }
}

// ------------------------------- host side ---------------------------------

static void gemm_dispatch(hipStream_t st, const GemmP& p) {
    dim3 grid((p.N + 255) / 256, (p.M + 31) / 32);
    if (p.kdiv == 1) {
        if (p.bAct == 1)      gemm_wmma_bf16<1, 1, 1, 1><<<grid, 256, 0, st>>>(p);
        else if (p.vecb)      gemm_wmma_bf16<1, 0, 0, 1><<<grid, 256, 0, st>>>(p);
        else                  gemm_wmma_bf16<1, 0, 0, 0><<<grid, 256, 0, st>>>(p);
    } else if (p.kdiv == 2) {
        if (p.outAct == 2) gemm_wmma_bf16<2, 0, 2, 0><<<grid, 256, 0, st>>>(p);
        else               gemm_wmma_bf16<2, 0, 3, 0><<<grid, 256, 0, st>>>(p);
    } else {
        if (p.bAct == 4) gemm_wmma_bf16<3, 4, 4, 0><<<grid, 256, 0, st>>>(p);
        else             gemm_wmma_bf16<3, 0, 0, 0><<<grid, 256, 0, st>>>(p);
    }
}

static void gemm_launch(hipStream_t st, const float* A, const float* B, float* D,
                        const float* bias, const float* addSrc,
                        int M, int K, int N, int aMs,
                        long bOff, long ks, int kdiv, int n2, int n3,
                        long s1, long s2, long s3,
                        int dil2, int off2, int bound2,
                        int dil3, int pad3, int bound3,
                        int bAct, int outAct, int accumulate, int vecb) {
    GemmP p;
    p.A = A; p.B = B; p.D = D; p.bias = bias; p.addSrc = addSrc;
    p.M = M; p.K = K; p.N = N; p.aMs = aMs;
    p.bOff = (int)bOff; p.ks = (int)ks; p.kdiv = kdiv;
    p.n2 = n2; p.n3 = n3; p.s1 = (int)s1; p.s2 = (int)s2; p.s3 = (int)s3;
    p.dil2 = dil2; p.off2 = off2; p.bound2 = bound2;
    p.dil3 = dil3; p.pad3 = pad3; p.bound3 = bound3;
    p.bAct = bAct; p.outAct = outAct; p.accumulate = accumulate;
    p.vecb = vecb;
    gemm_dispatch(st, p);
}

static void gemm_plain(hipStream_t st, const float* A, const float* B, float* D,
                       const float* bias, const float* addSrc,
                       int M, int K, int N, int aMs, int bAct, int outAct, int acc) {
    gemm_launch(st, A, B, D, bias, addSrc, M, K, N, aMs,
                0, (long)N, 1, 1, N, 0, 0, 1,
                0, 0, 1, 0, 0, N, bAct, outAct, acc, 1);
}

extern "C" void kernel_launch(void* const* d_in, const int* in_sizes, int n_in,
                              void* d_out, int out_size, void* d_ws, size_t ws_size,
                              hipStream_t stream) {
    (void)in_sizes; (void)n_in; (void)out_size; (void)ws_size;
    // Flattened setup_inputs() order.
    const float* x         = (const float*)d_in[0];
    const int*   stamp     = (const int*)d_in[1];
    const float* nodevec1  = (const float*)d_in[2];
    const float* nodevec2  = (const float*)d_in[3];
    const float* start_w   = (const float*)d_in[4];
    const float* start_b   = (const float*)d_in[5];
    // glayers: 6 + li*8 + {fw,fb,gw,gb,sw,sb,mw,mb}
    const float* end1_w    = (const float*)d_in[70];
    const float* end1_b    = (const float*)d_in[71];
    const float* end2_w    = (const float*)d_in[72];
    const float* end2_b    = (const float*)d_in[73];
    const float* reduce_w  = (const float*)d_in[74];
    const float* t1_w      = (const float*)d_in[75];
    const float* t1_b      = (const float*)d_in[76];
    const float* stamp_emb = (const float*)d_in[77];
    const float* pos_emb   = (const float*)d_in[78];
    const float* ln_g      = (const float*)d_in[79];
    const float* ln_b      = (const float*)d_in[80];
    const float* in_w      = (const float*)d_in[81];
    const float* in_b      = (const float*)d_in[82];
    // tblocks: 83 + bi*4 (bi<10); block 10 at 123..127 (w1,b1,w2,b2,proj)
    const float* reg_w     = (const float*)d_in[128];
    const float* reg_b     = (const float*)d_in[129];

    // -------- workspace arena (floats). Time and graph pools alias. --------
    float* ws  = (float*)d_ws;
    long   off = 0;
    float* ADP = ws + off; off += 512L * 512;
    float* T1  = ws + off; off += 128;
    float* YB  = ws + off; off += 12L * 16384;
    float* POOL = ws + off;
    // time-branch buffers
    float* TH0 = POOL;
    float* TH1 = TH0 + 320L * 16384;
    float* TC1 = TH1 + 320L * 16384;
    float* TRS = TC1 + 320L * 16384;
    // graph-branch buffers (alias POOL; branches run strictly sequentially)
    float* GA  = POOL;
    float* GB  = GA  + 32L * 32 * 13 * 512;
    float* GF  = GB  + 32L * 32 * 13 * 512;
    float* GAC = GF  + 32L * 32 * 12 * 512;
    float* GHA = GAC + 32L * 32 * 12 * 512;
    float* GHB = GHA + 32L * 32 * 12 * 512;
    float* GSK = GHB + 32L * 32 * 12 * 512;
    float* GE1 = GSK + 256L * 16384;

    // -------- small independent pieces --------
    t1_kernel<<<32, 16, 0, stream>>>(stamp, stamp_emb, pos_emb, ln_g, ln_b,
                                     reduce_w, t1_w, t1_b, T1);
    adp_kernel<<<512, 512, 0, stream>>>(nodevec1, nodevec2, ADP);

    // ============================ time branch ============================
    // H0[c, b, n] = sum_t in_w[c,t] * x[b,t,n,0] + in_b[c]   (strided B)
    gemm_launch(stream, in_w, x, TH0, in_b, nullptr,
                64, 12, 16384, 12,
                0, 1024, 1, 32, 512,
                0, 12288, 2,
                0, 0, 32, 0, 0, 512, 0, 0, 0, 0);

    float* Hc = TH0; float* Hn = TH1;
    for (int bi = 0; bi < 11; ++bi) {
        const int d = 1 << bi;
        const int O = (bi < 10) ? 64 : 320;
        const int base = (bi < 10) ? (83 + bi * 4) : 123;
        const float* w1 = (const float*)d_in[base + 0];
        const float* b1 = (const float*)d_in[base + 1];
        const float* w2 = (const float*)d_in[base + 2];
        const float* b2 = (const float*)d_in[base + 3];
        const float* res = Hc;
        if (bi == 10) {
            const float* proj = (const float*)d_in[127];
            gemm_plain(stream, proj, Hc, TRS, nullptr, nullptr,
                       320, 64, 16384, 64, 0, 0, 0);
            res = TRS;
        }
        // C1 = gelu( conv_d(gelu(Hc), w1) + b1 )   (input gelu in B-loader)
        gemm_launch(stream, w1, Hc, TC1, b1, nullptr,
                    O, 3 * 64, 16384, 3 * 64,
                    0, 16384, 3, 32, 512,
                    0, 512, 1,
                    0, 0, 32,
                    d, 1, 512,
                    4, 4, 0, 0);
        // Hn = conv_d(C1, w2) + b2 + res
        gemm_launch(stream, w2, TC1, Hn, b2, res,
                    O, 3 * O, 16384, 3 * O,
                    0, 16384, 3, 32, 512,
                    0, 512, 1,
                    0, 0, 32,
                    d, 1, 512,
                    0, 0, 0, 0);
        float* t = Hc; Hc = Hn; Hn = t;
    }
    // y1 = reg_w @ H + reg_b  -> [12, 16384] -> d_out first half
    gemm_plain(stream, reg_w, Hc, YB, reg_b, nullptr, 12, 320, 16384, 320, 0, 0, 0);
    ycopy_k<<<(196608 + 255) / 256, 256, 0, stream>>>((float*)d_out, YB);

    // =========================== graph branch ============================
    // g0[c,b,l,n] = start_w @ x(padded-left in time)  (strided B, elt stride 2)
    gemm_launch(stream, start_w, x, GA, start_b, nullptr,
                32, 2, 32 * 13 * 512, 2,
                0, 1, 1, 13, 512,
                12288, 1024, 2,
                0, 1, 12,
                0, 0, 512,
                0, 0, 0, 0);

    int Lin = 13;
    float* gc = GA; float* gn = GB;
    for (int li = 0; li < 8; ++li) {
        const int  d    = 1 << (li & 1);
        const int  Lout = Lin - d;
        const long NC   = 32L * Lout * 512;  // B * Lout * N columns
        const float* fw = (const float*)d_in[6 + li * 8 + 0];
        const float* fb = (const float*)d_in[6 + li * 8 + 1];
        const float* gw = (const float*)d_in[6 + li * 8 + 2];
        const float* gb = (const float*)d_in[6 + li * 8 + 3];
        const float* sw = (const float*)d_in[6 + li * 8 + 4];
        const float* sb = (const float*)d_in[6 + li * 8 + 5];
        const float* mw = (const float*)d_in[6 + li * 8 + 6];
        const float* mb = (const float*)d_in[6 + li * 8 + 7];

        // filt = tanh(tconv) ; gate = sigmoid(tconv) ; K = 2C taps interleaved
        gemm_launch(stream, fw, gc, GF, fb, nullptr,
                    32, 64, (int)NC, 64,
                    0, 32L * Lin * 512, 2, Lout, 512,
                    (long)Lin * 512, 512, 1,
                    d, 0, Lin,
                    0, 0, 512,
                    0, 2, 0, 0);
        gemm_launch(stream, gw, gc, GAC, gb, nullptr,
                    32, 64, (int)NC, 64,
                    0, 32L * Lin * 512, 2, Lout, 512,
                    (long)Lin * 512, 512, 1,
                    d, 0, Lin,
                    0, 0, 512,
                    0, 3, 0, 0);
        const long tot = 32 * NC;
        ew_mul_k<<<(int)((tot + 255) / 256), 256, 0, stream>>>(GF, GAC, tot);

        // skip: only the trailing time index survives -> accumulate
        // sw @ GF[:, :, Lout-1, :] into running [256, B*N] buffer.  (s3==1)
        gemm_launch(stream, sw, GF, GSK, sb, nullptr,
                    256, 32, 16384, 32,
                    (long)(Lout - 1) * 512, NC, 1, 32, 512,
                    0, (long)Lout * 512, 1,
                    0, 0, 32,
                    0, 0, 512,
                    0, 0, (li > 0) ? 1 : 0, 1);

        // gcn: ACC = t1_0*g + sum_k t1_k * (g @ ADP^k); hops as big WMMA GEMMs
        const long LN = (long)Lout * 512;
        const int  Mh = 32 * 32 * Lout;
        axpy_b_k<<<(int)((tot + 255) / 256), 256, 0, stream>>>(GAC, GF, T1, 0, LN, tot, 1);
        gemm_plain(stream, GF,  ADP, GHA, nullptr, nullptr, Mh, 512, 512, 512, 0, 0, 0);
        axpy_b_k<<<(int)((tot + 255) / 256), 256, 0, stream>>>(GAC, GHA, T1, 1, LN, tot, 0);
        gemm_plain(stream, GHA, ADP, GHB, nullptr, nullptr, Mh, 512, 512, 512, 0, 0, 0);
        axpy_b_k<<<(int)((tot + 255) / 256), 256, 0, stream>>>(GAC, GHB, T1, 2, LN, tot, 0);
        gemm_plain(stream, GHB, ADP, GHA, nullptr, nullptr, Mh, 512, 512, 512, 0, 0, 0);
        axpy_b_k<<<(int)((tot + 255) / 256), 256, 0, stream>>>(GAC, GHA, T1, 3, LN, tot, 0);

        // g_next = mw @ ACC + mb ; then residual from gc's trailing window
        gemm_plain(stream, mw, GAC, gn, mb, nullptr, 32, 32, (int)NC, 32, 0, 0, 0);
        add_res_k<<<(int)((tot + 255) / 256), 256, 0, stream>>>(gn, gc, Lout, Lin, d, tot);
        float* t = gc; gc = gn; gn = t;
        Lin = Lout;
    }

    // end head: relu(end1 @ relu(skip) + b1) ; end2 ; transpose to y2 slot
    gemm_plain(stream, end1_w, GSK, GE1, end1_b, nullptr, 512, 256, 16384, 256, 1, 1, 0);
    gemm_plain(stream, end2_w, GE1, YB,  end2_b, nullptr, 12, 512, 16384, 512, 0, 0, 0);
    ycopy_k<<<(196608 + 255) / 256, 256, 0, stream>>>((float*)d_out + 196608, YB);
}
